// Attention_14791867368086
// MI455X (gfx1250) — compile-verified, matrix operand order checked by reading, not run on hardware
//
#include <hip/hip_runtime.h>
#include <hip/hip_bf16.h>

// ---------- WMMA types (gfx1250, wave32) ----------
typedef __attribute__((ext_vector_type(16))) __bf16 v16bf;
typedef __attribute__((ext_vector_type(8)))  float  v8f;

struct __align__(16) U128 { unsigned int w[4]; };

union BfFrag {
    v16bf v;
    U128  q[2];
    unsigned short u[16];
};

__device__ __forceinline__ unsigned short f2bf(float f) {
    unsigned int u = __float_as_uint(f);
    unsigned int r = u + 0x7fffu + ((u >> 16) & 1u);   // round-to-nearest-even
    return (unsigned short)(r >> 16);
}

__device__ __forceinline__ v8f v8f_zero() {
    v8f z;
    #pragma unroll
    for (int i = 0; i < 8; i++) z[i] = 0.0f;
    return z;
}

#define WMMA_BF16(a, b, c) \
    __builtin_amdgcn_wmma_f32_16x16x32_bf16(false, (a), false, (b), (short)0, (c), false, false)

// ---------- CDNA5 async global->LDS (ASYNCcnt path, no VGPR round trip) ----------
// Generic pointers to __shared__ carry the LDS byte offset in their low 32 bits
// (ISA 10.2: LDS aperture).
__device__ __forceinline__ void async_load_lds_b128(void* lds_ptr, const void* gaddr) {
    unsigned lds_off = (unsigned)(unsigned long long)lds_ptr;
    asm volatile("global_load_async_to_lds_b128 %0, %1, off"
                 :: "v"(lds_off), "v"(gaddr)
                 : "memory");
}
__device__ __forceinline__ void wait_async0() {
    asm volatile("s_wait_asynccnt 0x0" ::: "memory");
}

// ---------- fp32 -> bf16 conversion (linear) ----------
__global__ void cvt_f32_bf16(const float* __restrict__ src,
                             unsigned short* __restrict__ dst, int n) {
    int i = blockIdx.x * blockDim.x + threadIdx.x;
    int stride = gridDim.x * blockDim.x;
    for (; i < n; i += stride) dst[i] = f2bf(src[i]);
}

// ---------- fp32 (K x N) -> bf16 transposed (N x K) ----------
__global__ void cvt_transpose_bf16(const float* __restrict__ src,
                                   unsigned short* __restrict__ dst,
                                   int K, int N) {
    int i = blockIdx.x * blockDim.x + threadIdx.x;
    int stride = gridDim.x * blockDim.x;
    int total = K * N;
    for (; i < total; i += stride) {
        int k = i / N, n = i - k * N;          // coalesced read along n
        dst[(size_t)n * K + k] = f2bf(src[i]);
    }
}

// ---------- WMMA GEMM: C(MxN) = A(MxK) * Bt(NxK)^T, bf16 in, f32 acc ----------
// Block: 256 threads = 8 waves (2m x 4n). Macro tile 128(M) x 256(N).
// Wave tile 64x64 = 16 accumulators. K-step 32.
// Double-buffered LDS filled by async global->LDS; one barrier per k-step.
template <bool STORE_F32>
__global__ __launch_bounds__(256) void gemm_bf16_wmma(
    const unsigned short* __restrict__ A,    // M x K
    const unsigned short* __restrict__ Bt,   // N x K (pre-transposed)
    void* __restrict__ C, int M, int N, int K)
{
    __shared__ __align__(16) unsigned short lA[2][128 * 40];
    __shared__ __align__(16) unsigned short lB[2][256 * 40];

    const int tid  = threadIdx.x;
    const int lane = tid & 31;
    const int wave = tid >> 5;
    const int wm   = wave & 1;        // 2 m-subtiles of 64
    const int wn   = wave >> 1;       // 4 n-subtiles of 64
    const int lr   = lane & 15;
    const int g    = lane >> 4;
    const int m0   = blockIdx.y * 128;
    const int n0   = blockIdx.x * 256;

    // staging coords: thread -> (row, 8-elem k-group)
    const int kq = tid & 3;           // elements kq*8 .. kq*8+7 of the 32-k tile
    const int r0 = tid >> 2;          // base row (0..63), extended by +64 strides
    const unsigned short* pa = A  + (size_t)(m0 + r0) * K + kq * 8;
    const unsigned short* pb = Bt + (size_t)(n0 + r0) * K + kq * 8;

    v8f acc[4][4];
    #pragma unroll
    for (int a = 0; a < 4; a++)
        #pragma unroll
        for (int b = 0; b < 4; b++) acc[a][b] = v8f_zero();

    const int nk = K >> 5;

    // async-stage one 32-k tile into LDS buffer `buf`
    auto stage_tile = [&](int kt, int buf) {
        const size_t ko = (size_t)kt * 32;
        #pragma unroll
        for (int i = 0; i < 2; i++)
            async_load_lds_b128(&lA[buf][(r0 + i * 64) * 40 + kq * 8],
                                pa + ko + (size_t)i * 64 * K);
        #pragma unroll
        for (int j = 0; j < 4; j++)
            async_load_lds_b128(&lB[buf][(r0 + j * 64) * 40 + kq * 8],
                                pb + ko + (size_t)j * 64 * K);
    };

    // prologue
    stage_tile(0, 0);
    wait_async0();
    __syncthreads();

    for (int kt = 0; kt < nk; kt++) {
        const int cur = kt & 1, nxt = cur ^ 1;

        // issue next tile's async loads; safe: end-of-iteration barrier of kt-1
        // guarantees no wave still reads buffer `nxt`
        if (kt + 1 < nk) stage_tile(kt + 1, nxt);
        // L2 prefetch two tiles ahead
        if (kt + 2 < nk) {
            const size_t ko2 = (size_t)(kt + 2) * 32;
            __builtin_prefetch(pa + ko2, 0, 1);
            __builtin_prefetch(pb + ko2, 0, 1);
        }

        // fragments from current buffer
        BfFrag af[4], bf[4];
        #pragma unroll
        for (int mt = 0; mt < 4; mt++) {
            const unsigned short* p = &lA[cur][(wm * 64 + mt * 16 + lr) * 40 + g * 8];
            af[mt].q[0] = *(const U128*)p;
            af[mt].q[1] = *(const U128*)(p + 16);
        }
        #pragma unroll
        for (int nt = 0; nt < 4; nt++) {
            const unsigned short* p = &lB[cur][(wn * 64 + nt * 16 + lr) * 40 + g * 8];
            bf[nt].q[0] = *(const U128*)p;
            bf[nt].q[1] = *(const U128*)(p + 16);
        }
        #pragma unroll
        for (int mt = 0; mt < 4; mt++)
            #pragma unroll
            for (int nt = 0; nt < 4; nt++)
                acc[mt][nt] = WMMA_BF16(af[mt].v, bf[nt].v, acc[mt][nt]);

        wait_async0();       // my async writes into `nxt` complete
        __syncthreads();     // everyone's complete & visible
    }

    // C layout: VGPR vr -> row (lanes 0-15: M=vr, lanes 16-31: M=vr+8), lane%16 -> col
    #pragma unroll
    for (int mt = 0; mt < 4; mt++)
        #pragma unroll
        for (int nt = 0; nt < 4; nt++)
            #pragma unroll
            for (int vr = 0; vr < 8; vr++) {
                int row = m0 + wm * 64 + mt * 16 + vr + g * 8;
                int col = n0 + wn * 64 + nt * 16 + lr;
                float val = acc[mt][nt][vr];
                if (STORE_F32)
                    ((float*)C)[(size_t)row * N + col] = val;
                else
                    ((unsigned short*)C)[(size_t)row * N + col] = f2bf(val);
            }
}

// ---------- Flash attention (causal), bf16 WMMA, fp32 online softmax ----------
// Grid: (T/128, B*H). Block: 256 thr = 8 waves; wave owns 16 q rows.
__global__ __launch_bounds__(256) void flash_attn_wmma(
    const unsigned short* __restrict__ qkv,  // (B*T, 3*D) bf16, Q|K|V
    unsigned short* __restrict__ ctx,        // (B*T, D) bf16
    int T, int H, int Dh)                    // 2048, 16, 64
{
    const int D  = H * Dh;     // 1024
    const int W3 = 3 * D;      // 3072

    __shared__ __align__(16) unsigned short lK[64 * 72];      // [key][d]
    __shared__ __align__(16) unsigned short lV[64 * 72];      // [d][key] (transposed)
    __shared__ __align__(16) unsigned short lP[8][16 * 72];   // per-wave P buffer

    const int tid  = threadIdx.x;
    const int lane = tid & 31, wave = tid >> 5;
    const int lr   = lane & 15, g = lane >> 4;
    const int bh   = blockIdx.y;
    const int b    = bh / H, h = bh % H;
    const int qbase = blockIdx.x * 128;
    const int qrow0 = qbase + wave * 16;

    // Q fragments: A-matrix 16x64 over d, split into two K=32 fragments
    BfFrag qf[2];
    {
        const size_t rq = (size_t)(b * T + qrow0 + lr) * W3 + (size_t)h * Dh;
        #pragma unroll
        for (int db = 0; db < 2; db++) {
            const unsigned short* p = &qkv[rq + db * 32 + g * 8];
            qf[db].q[0] = *(const U128*)p;
            qf[db].q[1] = *(const U128*)(p + 16);
        }
    }

    v8f oacc[4];
    float m[8], l[8];
    #pragma unroll
    for (int dt = 0; dt < 4; dt++) oacc[dt] = v8f_zero();
    #pragma unroll
    for (int vr = 0; vr < 8; vr++) { m[vr] = -3e38f; l[vr] = 0.0f; }

    const int nkt = (qbase + 128) / 64;       // causal tile bound for this block
    const int myMaxRow = qrow0 + 15;

    for (int kt = 0; kt < nkt; kt++) {
        __syncthreads();                      // previous tile's readers done
        // K tile via async global->LDS: 512 b128 slots, 2 per thread
        #pragma unroll
        for (int i = 0; i < 2; i++) {
            int e   = tid + i * 256;
            int key = e >> 3, dq = e & 7;
            const unsigned short* gp =
                qkv + (size_t)(b * T + kt * 64 + key) * W3 + D + (size_t)h * Dh + dq * 8;
            async_load_lds_b128(&lK[key * 72 + dq * 8], gp);
        }
        // V tile transposed [d][key]: manual scatter (async cannot transpose)
        #pragma unroll
        for (int i = 0; i < 16; i++) {
            int e = tid + i * 256;
            int key = e >> 6, d = e & 63;
            lV[d * 72 + key] =
                qkv[(size_t)(b * T + kt * 64 + key) * W3 + 2 * D + (size_t)h * Dh + d];
        }
        wait_async0();
        __syncthreads();
        if (kt * 64 > myMaxRow) continue;     // wave fully above diagonal

        // S = Q K^T : 4 col-tiles of 16 keys, contraction over d=64
        v8f s[4];
        #pragma unroll
        for (int ct = 0; ct < 4; ct++) {
            v8f z = v8f_zero();
            #pragma unroll
            for (int db = 0; db < 2; db++) {
                BfFrag kf;
                const unsigned short* p = &lK[(ct * 16 + lr) * 72 + db * 32 + g * 8];
                kf.q[0] = *(const U128*)p;
                kf.q[1] = *(const U128*)(p + 16);
                z = WMMA_BF16(qf[db].v, kf.v, z);
            }
            s[ct] = z;
        }

        // scale + causal mask
        #pragma unroll
        for (int ct = 0; ct < 4; ct++) {
            int col = kt * 64 + ct * 16 + lr;
            #pragma unroll
            for (int vr = 0; vr < 8; vr++) {
                int row = qrow0 + vr + g * 8;
                float v = s[ct][vr] * 0.125f;            // 1/sqrt(64)
                if (col > row) v = -1e30f;
                s[ct][vr] = v;
            }
        }

        // online softmax: row stats live per-lane over its 8 rows
        float corr[8];
        #pragma unroll
        for (int vr = 0; vr < 8; vr++) {
            float x = fmaxf(fmaxf(s[0][vr], s[1][vr]), fmaxf(s[2][vr], s[3][vr]));
            #pragma unroll
            for (int msk = 1; msk < 16; msk <<= 1)
                x = fmaxf(x, __shfl_xor(x, msk, 16));
            float mn = fmaxf(m[vr], x);
            corr[vr] = __expf(m[vr] - mn);
            m[vr] = mn;
        }
        #pragma unroll
        for (int ct = 0; ct < 4; ct++)
            #pragma unroll
            for (int vr = 0; vr < 8; vr++)
                s[ct][vr] = __expf(s[ct][vr] - m[vr]);
        #pragma unroll
        for (int vr = 0; vr < 8; vr++) {
            float rs = s[0][vr] + s[1][vr] + s[2][vr] + s[3][vr];
            #pragma unroll
            for (int msk = 1; msk < 16; msk <<= 1)
                rs += __shfl_xor(rs, msk, 16);
            l[vr] = l[vr] * corr[vr] + rs;
        }
        #pragma unroll
        for (int dt = 0; dt < 4; dt++)
            #pragma unroll
            for (int vr = 0; vr < 8; vr++)
                oacc[dt][vr] *= corr[vr];

        // C-layout P -> bf16 A-layout via per-wave LDS round trip
        #pragma unroll
        for (int ct = 0; ct < 4; ct++)
            #pragma unroll
            for (int vr = 0; vr < 8; vr++)
                lP[wave][(vr + g * 8) * 72 + ct * 16 + lr] = f2bf(s[ct][vr]);

        BfFrag pf[2];
        #pragma unroll
        for (int kb = 0; kb < 2; kb++) {
            const unsigned short* p = &lP[wave][lr * 72 + kb * 32 + g * 8];
            pf[kb].q[0] = *(const U128*)p;
            pf[kb].q[1] = *(const U128*)(p + 16);
        }

        // O += P V : 4 d-tiles of 16, contraction over 64 keys
        #pragma unroll
        for (int dt = 0; dt < 4; dt++) {
            v8f z = oacc[dt];
            #pragma unroll
            for (int kb = 0; kb < 2; kb++) {
                BfFrag vf;
                const unsigned short* p = &lV[(dt * 16 + lr) * 72 + kb * 32 + g * 8];
                vf.q[0] = *(const U128*)p;
                vf.q[1] = *(const U128*)(p + 16);
                z = WMMA_BF16(pf[kb].v, vf.v, z);
            }
            oacc[dt] = z;
        }
    }

    // normalize and store context (B*T, D) bf16
    #pragma unroll
    for (int vr = 0; vr < 8; vr++) {
        float inv = (l[vr] > 0.0f) ? (1.0f / l[vr]) : 0.0f;
        size_t row = (size_t)(b * T + qrow0 + vr + g * 8);
        #pragma unroll
        for (int dt = 0; dt < 4; dt++)
            ctx[row * D + (size_t)h * Dh + dt * 16 + lr] = f2bf(oacc[dt][vr] * inv);
    }
}

// ---------- launcher ----------
extern "C" void kernel_launch(void* const* d_in, const int* in_sizes, int n_in,
                              void* d_out, int out_size, void* d_ws, size_t ws_size,
                              hipStream_t stream) {
    (void)in_sizes; (void)n_in; (void)out_size; (void)ws_size;

    const float* x      = (const float*)d_in[0];
    const float* w_qkv  = (const float*)d_in[1];
    const float* w_proj = (const float*)d_in[2];

    const int B = 4, T = 2048, D = 1024, H = 16, Dh = 64;
    const int M  = B * T;    // 8192
    const int N1 = 3 * D;    // 3072

    // workspace layout (bf16 buffers), all 16B-aligned
    unsigned short* xb   = (unsigned short*)d_ws;                 // x bf16 [M][K]      (16 MB)
    unsigned short* wqb  = xb   + (size_t)M * D;                  // w_qkv^T [N1][D]    ( 6 MB)
    unsigned short* wpb  = wqb  + (size_t)D * N1;                 // w_proj^T [D][D]    ( 2 MB)
    unsigned short* qkvb = wpb  + (size_t)D * D;                  // qkv bf16 [M][N1]   (48 MB)
    unsigned short* ctxb = qkvb + (size_t)M * N1;                 // ctx bf16 [M][D]    (16 MB)

    cvt_f32_bf16<<<1024, 256, 0, stream>>>(x, xb, M * D);
    cvt_transpose_bf16<<<1024, 256, 0, stream>>>(w_qkv, wqb, D, N1);  // (D x 3D) -> (3D x D)
    cvt_transpose_bf16<<<1024, 256, 0, stream>>>(w_proj, wpb, D, D);  // (D x D)  -> (D x D)

    dim3 g1(N1 / 256, M / 128);
    gemm_bf16_wmma<false><<<g1, 256, 0, stream>>>(xb, wqb, (void*)qkvb, M, N1, D);

    dim3 g2(T / 128, B * H);
    flash_attn_wmma<<<g2, 256, 0, stream>>>(qkvb, ctxb, T, H, Dh);

    dim3 g3(D / 256, M / 128);
    gemm_bf16_wmma<true><<<g3, 256, 0, stream>>>(ctxb, wpb, d_out, M, D, D);
}
